// DND_9569187136125
// MI455X (gfx1250) — matrix operand order repaired and checked by reading, not compiled, not run on hardware
//
#include <hip/hip_runtime.h>
#include <hip/hip_bf16.h>

// ---------------------------------------------------------------------------
// IDW-kNN (B=256 queries, N=131072 keys, D=128, K=50) for MI455X / gfx1250.
//
// Pass 1: per (16-query block, 4096-key chunk) workgroup:
//   - split-precision bf16 WMMA GEMM (hi/lo decomposition, 3 WMMAs per k-step)
//     -> fp32-accurate squared distances
//   - per-thread sorted top-50 lists in LDS (8 threads per query)
//   - 8-way merge -> 50 (dist, value) candidates per (query, chunk) in ws
// Pass 2: one wave per query; lane l owns sorted list of chunk l (32 chunks =
//   32 lanes); 32-way merge via shfl_xor min-reduction; accumulate
//   sum(w), sum(w*v) over the global top-50; out = sum(w*v)/sum(w).
// ---------------------------------------------------------------------------

#define DELTA_K 1e-3f
#define BQ 256
#define NK 131072
#define DIM 128
#define KNB 50

#define CHUNKS 32
#define COLS_PER_CHUNK 4096      // NK / CHUNKS
#define COLS_PER_ROUND 64        // 4 waves * 16 cols
#define ROUNDS 64                // COLS_PER_CHUNK / COLS_PER_ROUND
#define THREADS_P1 128           // 4 waves (wave32)
#define SUBS 8                   // threads per query in the selection stage

typedef __attribute__((ext_vector_type(16))) __bf16 v16bf;
typedef __attribute__((ext_vector_type(8)))  float  v8f;

__global__ __launch_bounds__(THREADS_P1)
void knn_pass1(const float* __restrict__ qry,   // [BQ, DIM]
               const float* __restrict__ keys,  // [NK, DIM]
               const float* __restrict__ vals,  // [NK]
               float* __restrict__ wsd,         // [BQ][CHUNKS][KNB]
               float* __restrict__ wsv)         // [BQ][CHUNKS][KNB]
{
    __shared__ __bf16 qhi[16 * DIM];
    __shared__ __bf16 qlo[16 * DIM];
    __shared__ float  qsq[16];
    __shared__ float  stage_d[16 * COLS_PER_ROUND];
    __shared__ float  stage_v[COLS_PER_ROUND];
    __shared__ float  list_d[THREADS_P1 * KNB];
    __shared__ float  list_v[THREADS_P1 * KNB];

    const int tid   = threadIdx.x;
    const int lane  = tid & 31;
    const int wv    = tid >> 5;                 // wave id 0..3
    const int q0    = blockIdx.y * 16;          // query block base
    const int chunk = blockIdx.x;
    const int nbase = chunk * COLS_PER_CHUNK;

    // ---- load queries, split fp32 -> bf16 hi + bf16 lo ----
    for (int i = tid; i < 16 * DIM; i += THREADS_P1) {
        float  x = qry[q0 * DIM + i];
        __bf16 h = (__bf16)x;
        qhi[i] = h;
        qlo[i] = (__bf16)(x - (float)h);
    }
    for (int i = tid; i < THREADS_P1 * KNB; i += THREADS_P1) {
        list_d[i] = 3.402823e38f;
        list_v[i] = 0.0f;
    }
    __syncthreads();
    if (tid < 16) {
        float s = 0.0f;
        for (int k = 0; k < DIM; ++k) {
            float x = qry[(q0 + tid) * DIM + k];
            s += x * x;
        }
        qsq[tid] = s;
    }
    __syncthreads();

    // ---- A fragments (16x32 bf16 per k-step; layout: lanes 0-15 hold
    //      K {0..15}, lanes 16-31 hold K {16..31} of row lane&15)
    v16bf ah[4], al[4];
    {
        const int r    = lane & 15;
        const int koff = (lane >> 4) << 4;
#pragma unroll
        for (int f = 0; f < 4; ++f) {
            const int kb = f * 32 + koff;
#pragma unroll
            for (int e = 0; e < 8; ++e) {
                ah[f][e]     = qhi[r * DIM + kb + e];
                al[f][e]     = qlo[r * DIM + kb + e];
                ah[f][8 + e] = qhi[r * DIM + kb + 8 + e];
                al[f][8 + e] = qlo[r * DIM + kb + 8 + e];
            }
        }
    }

    const int mb = (lane >> 4) << 3;   // C layout: M offset for this lane half
    const int nl = lane & 15;          // C layout: local column

    for (int r = 0; r < ROUNDS; ++r) {
        const int col0 = nbase + r * COLS_PER_ROUND + wv * 16;
        const int row  = col0 + nl;                 // key index this lane owns
        const int kb16 = (lane >> 4) << 4;          // B: contiguous 16-K half

        if (r + 1 < ROUNDS) {
            // warm L2/WGP$ for next round's key rows (global_prefetch_b8)
            __builtin_prefetch(keys + (size_t)(row + COLS_PER_ROUND) * DIM + kb16, 0, 1);
        }

        v8f   acc = {};
        float sq  = 0.0f;
#pragma unroll
        for (int f = 0; f < 4; ++f) {
            v16bf bh, bl;
            const float* src = keys + (size_t)row * DIM + f * 32 + kb16;
#pragma unroll
            for (int e = 0; e < 16; ++e) {
                float  x = src[e];
                sq += x * x;
                __bf16 h = (__bf16)x;
                bh[e] = h;
                bl[e] = (__bf16)(x - (float)h);
            }
            // cross ~= Alo*Bhi + Ahi*Blo + Ahi*Bhi  (near-fp32 dot product)
            acc = __builtin_amdgcn_wmma_f32_16x16x32_bf16(false, al[f], false, bh,
                                                          (short)0, acc, false, false);
            acc = __builtin_amdgcn_wmma_f32_16x16x32_bf16(false, ah[f], false, bl,
                                                          (short)0, acc, false, false);
            acc = __builtin_amdgcn_wmma_f32_16x16x32_bf16(false, ah[f], false, bh,
                                                          (short)0, acc, false, false);
        }
        // lanes L and L+16 loaded complementary K-halves of the same key row
        float dsq  = sq + __shfl_xor(sq, 16, 32);
        float vcol = vals[row];
        if (lane < 16) stage_v[wv * 16 + lane] = vcol;

#pragma unroll
        for (int j = 0; j < 8; ++j) {
            float d2 = qsq[mb + j] + dsq - 2.0f * acc[j];
            d2 = d2 < 0.0f ? 0.0f : d2;
            stage_d[(mb + j) * COLS_PER_ROUND + wv * 16 + nl] = d2;
        }
        __syncthreads();

        // ---- selection: thread t owns (query t>>3, sub-stream t&7) ----
        {
            const int q    = tid >> 3;
            const int sub  = tid & 7;
            const int base = tid * KNB;
#pragma unroll
            for (int cc = 0; cc < COLS_PER_ROUND / 8; ++cc) {
                const int   c = sub + cc * 8;
                const float d = stage_d[q * COLS_PER_ROUND + c];
                if (d < list_d[base + KNB - 1]) {
                    const float v = stage_v[c];
                    int pos = KNB - 1;
                    while (pos > 0 && list_d[base + pos - 1] > d) {
                        list_d[base + pos] = list_d[base + pos - 1];
                        list_v[base + pos] = list_v[base + pos - 1];
                        --pos;
                    }
                    list_d[base + pos] = d;
                    list_v[base + pos] = v;
                }
            }
        }
        __syncthreads();
    }

    // ---- 8-way merge of sorted sub-lists -> 50 candidates per query ----
    if (tid < 16) {
        int heads[SUBS];
#pragma unroll
        for (int s = 0; s < SUBS; ++s) heads[s] = 0;
        const int ob = ((q0 + tid) * CHUNKS + chunk) * KNB;
        for (int k = 0; k < KNB; ++k) {
            float best = 3.402823e38f;
            int   bs   = 0;
#pragma unroll
            for (int s = 0; s < SUBS; ++s) {
                const int   h  = heads[s];
                const float dd = (h < KNB) ? list_d[(tid * SUBS + s) * KNB + h]
                                           : 3.402823e38f;
                if (dd < best) { best = dd; bs = s; }
            }
            wsd[ob + k] = best;
            wsv[ob + k] = list_v[(tid * SUBS + bs) * KNB + heads[bs]];
            heads[bs]++;
        }
    }
}

__global__ __launch_bounds__(32)
void knn_pass2(const float* __restrict__ wsd, const float* __restrict__ wsv,
               float* __restrict__ out)
{
    const int q    = blockIdx.x;
    const int lane = threadIdx.x;                 // 32 lanes == 32 chunk lists
    const float* dl = wsd + ((size_t)q * CHUNKS + lane) * KNB;
    const float* vl = wsv + ((size_t)q * CHUNKS + lane) * KNB;

    int   p  = 0;
    float hd = dl[0];
    float hv = vl[0];
    float sw = 0.0f, swv = 0.0f;

    for (int k = 0; k < KNB; ++k) {
        float d   = hd;
        int   who = lane;
#pragma unroll
        for (int off = 16; off >= 1; off >>= 1) {
            const float od = __shfl_xor(d, off, 32);
            const int   ow = __shfl_xor(who, off, 32);
            if (od < d || (od == d && ow < who)) { d = od; who = ow; }
        }
        const float v = __shfl(hv, who, 32);
        const float w = 1.0f / (d + DELTA_K);
        sw  += w;
        swv += w * v;
        if (lane == who) {
            ++p;
            hd = (p < KNB) ? dl[p] : 3.402823e38f;
            hv = (p < KNB) ? vl[p] : 0.0f;
        }
    }
    if (lane == 0) out[q] = swv / sw;
}

extern "C" void kernel_launch(void* const* d_in, const int* in_sizes, int n_in,
                              void* d_out, int out_size, void* d_ws, size_t ws_size,
                              hipStream_t stream)
{
    const float* qry  = (const float*)d_in[0];   // [256,128] fp32
    const float* keys = (const float*)d_in[1];   // [131072,128] fp32
    const float* vals = (const float*)d_in[2];   // [131072,1] fp32
    (void)in_sizes; (void)n_in; (void)out_size; (void)ws_size;

    float* wsd = (float*)d_ws;                              // dists
    float* wsv = wsd + (size_t)BQ * CHUNKS * KNB;           // values

    knn_pass1<<<dim3(CHUNKS, BQ / 16), THREADS_P1, 0, stream>>>(qry, keys, vals, wsd, wsv);
    knn_pass2<<<BQ, 32, 0, stream>>>(wsd, wsv, (float*)d_out);
}
